// Ho_encoder_36155034698034
// MI455X (gfx1250) — matrix operand order repaired and checked by reading, not compiled, not run on hardware
//
#include <hip/hip_runtime.h>
#include <stdint.h>

// ---------------- problem constants (fixed by the reference) ----------------
#define NNODES 50000
#define HDIM   256
#define NPATH  4
#define NEDGE  800000
#define EPSV   1e-8f

// LDS pitch (in ushorts) for staged B matrix: 264 -> 132 dwords -> bank stride 4,
// so lanes 0..15 reading different rows hit distinct banks.
#define BPITCH 264

typedef __bf16 v16bf __attribute__((ext_vector_type(16)));
typedef __bf16 v8bf  __attribute__((ext_vector_type(8)));
typedef float  v8f   __attribute__((ext_vector_type(8)));
typedef float  v4f   __attribute__((ext_vector_type(4)));

__device__ __forceinline__ unsigned short f2bf_raw(float f) {
  unsigned u = __builtin_bit_cast(unsigned, f);
  u += 0x7FFFu + ((u >> 16) & 1u);           // round-to-nearest-even
  return (unsigned short)(u >> 16);
}
__device__ __forceinline__ float bf2f(unsigned short u) {
  unsigned x = ((unsigned)u) << 16;
  return __builtin_bit_cast(float, x);
}

// Native no-return f32 atomic add (STOREcnt-tracked); avoids any CAS expansion.
__device__ __forceinline__ void global_fadd(float* p, float v) {
  asm volatile("global_atomic_add_f32 %0, %1, off" :: "v"(p), "v"(v) : "memory");
}

// Async global->LDS 16B copy (ASYNCcnt-tracked), CDNA5 path: data never touches VGPRs.
__device__ __forceinline__ void async_ld16_to_lds(unsigned lds_byte, const void* gaddr) {
  asm volatile("global_load_async_to_lds_b128 %0, %1, off"
               :: "v"(lds_byte), "v"(gaddr) : "memory");
}
__device__ __forceinline__ void wait_async0() {
  asm volatile("s_wait_asynccnt 0x0" ::: "memory");
}

// ---------------- utility kernels ----------------
__global__ void __launch_bounds__(256) k_zero_f4(float* __restrict__ p, int n4) {
  int i = blockIdx.x * 256 + threadIdx.x;
  if (i < n4) { v4f z = {0.f, 0.f, 0.f, 0.f}; ((v4f*)p)[i] = z; }
}

__global__ void __launch_bounds__(256) k_f32_to_bf16(const float* __restrict__ src,
                                                     unsigned short* __restrict__ dst,
                                                     int n4) {
  int i = blockIdx.x * 256 + threadIdx.x;
  if (i >= n4) return;
  v4f f = ((const v4f*)src)[i];
  ushort4 o;
  o.x = f2bf_raw(f[0]); o.y = f2bf_raw(f[1]);
  o.z = f2bf_raw(f[2]); o.w = f2bf_raw(f[3]);
  ((ushort4*)dst)[i] = o;
}

// Stage a 256x256 bf16 matrix (row-major, k contiguous) into padded LDS via
// async global->LDS copies. All 256 threads participate; 16B per issue.
__device__ __forceinline__ void stage_B_async(unsigned short* smem,
                                              const unsigned short* __restrict__ src,
                                              int tid) {
  for (int c = tid; c < (HDIM * HDIM) / 8; c += 256) {
    int j = c >> 5;
    int k = (c & 31) << 3;
    unsigned lds_byte = (unsigned)(uintptr_t)(smem + j * BPITCH + k);
    async_ld16_to_lds(lds_byte, (const void*)(src + j * HDIM + k));
  }
  wait_async0();
  __syncthreads();
}

// ---------------- GEMM 1: X(bf16) = h @ W_p^T  (bf16 WMMA, f32 accum) ----------
// Block = 256 threads = 8 waves, covers 64 rows x 256 cols.
// Wave w: rows [blk*64 + (w&3)*16, +16), cols [(w>>2)*128, +128) = 8 j-tiles.
__global__ void __launch_bounds__(256) k_gemm_xw(const unsigned short* __restrict__ hbf,
                                                 const unsigned short* __restrict__ wbf,
                                                 unsigned short* __restrict__ X16) {
  extern __shared__ unsigned short smem[];   // HDIM x BPITCH ushorts (132 KB)
  const int tid = threadIdx.x;
  stage_B_async(smem, wbf, tid);

  const int lane = tid & 31;
  const int wave = tid >> 5;
  const int hl   = lane >> 4;        // half-wave select
  const int l16  = lane & 15;
  const int rowBase = blockIdx.x * 64 + (wave & 3) * 16;
  const int jBase   = (wave >> 2) * 128;
  const int r = rowBase + l16;

  v8f C[8];
  #pragma unroll
  for (int t = 0; t < 8; ++t) __builtin_memset(&C[t], 0, sizeof(v8f));

  for (int kc = 0; kc < HDIM; kc += 32) {
    // A fragment: lane holds row r; lanes<16: K {kc..kc+7, kc+16..kc+23},
    // lanes>=16: K {kc+8..kc+15, kc+24..kc+31}.
    v16bf a;
    if (r < NNODES) {
      const unsigned short* ap = hbf + (size_t)r * HDIM + kc + hl * 8;
      v8bf lo = *(const v8bf*)ap;
      v8bf hi = *(const v8bf*)(ap + 16);
      a = __builtin_shufflevector(lo, hi, 0,1,2,3,4,5,6,7,8,9,10,11,12,13,14,15);
    } else {
      __builtin_memset(&a, 0, sizeof(a));
    }
    #pragma unroll
    for (int jt = 0; jt < 8; ++jt) {
      // B fragment: lane holds col jBase+jt*16+l16, K {kc+hl*8..+7, kc+16+hl*8..+7}.
      const unsigned short* bp = smem + (jBase + jt * 16 + l16) * BPITCH + kc + hl * 8;
      v8bf blo = *(const v8bf*)bp;
      v8bf bhi = *(const v8bf*)(bp + 16);
      v16bf b = __builtin_shufflevector(blo, bhi, 0,1,2,3,4,5,6,7,8,9,10,11,12,13,14,15);
      C[jt] = __builtin_amdgcn_wmma_f32_16x16x32_bf16(false, a, false, b,
                                                      (short)0, C[jt], false, false);
    }
  }

  // C/D layout: lanes 0-15 VGPR v -> M=v; lanes 16-31 VGPR v -> M=8+v; N=lane%16.
  // Store X as bf16: halves the random-gather traffic in the scatter stage.
  #pragma unroll
  for (int jt = 0; jt < 8; ++jt) {
    const int col = jBase + jt * 16 + l16;
    #pragma unroll
    for (int v = 0; v < 8; ++v) {
      int rr = rowBase + v + hl * 8;
      if (rr < NNODES) X16[(size_t)rr * HDIM + col] = f2bf_raw(C[jt][v]);
    }
  }
}

// ---------------- edge scatter: emb[rows[e]] += vals[e]*X[cols[e]] -------------
__global__ void __launch_bounds__(256) k_scatter(const int* __restrict__ rows,
                                                 const int* __restrict__ cols,
                                                 const float* __restrict__ vals,
                                                 const unsigned short* __restrict__ X16,
                                                 float* __restrict__ emb,
                                                 float* __restrict__ deg) {
  long gid = (long)blockIdx.x * 256 + threadIdx.x;
  int e = (int)(gid >> 6);              // 64 threads per edge (4 values each)
  int q = (int)(gid & 63);
  if (e >= NEDGE) return;
  int rr = rows[e];
  int cc = cols[e];
  float v = vals[e];
  ushort4 x = *(const ushort4*)(X16 + (size_t)cc * HDIM + q * 4);
  float* base = emb + (size_t)rr * HDIM + q * 4;
  global_fadd(base + 0, v * bf2f(x.x));
  global_fadd(base + 1, v * bf2f(x.y));
  global_fadd(base + 2, v * bf2f(x.z));
  global_fadd(base + 3, v * bf2f(x.w));
  if (q == 0) global_fadd(deg + rr, v);
}

// ---------------- normalize + PReLU (in place over all paths) -----------------
__global__ void __launch_bounds__(256) k_norm_prelu(float* __restrict__ emb,
                                                    const float* __restrict__ deg,
                                                    const float* __restrict__ prelu_a) {
  long gid = (long)blockIdx.x * 256 + threadIdx.x;   // float4 index
  const long total4 = (long)NPATH * NNODES * HDIM / 4;
  if (gid >= total4) return;
  long g = gid * 4;
  int p = (int)(g / ((long)NNODES * HDIM));
  long rem = g - (long)p * NNODES * HDIM;
  int n = (int)(rem >> 8);                           // HDIM == 256
  float d = deg[(long)p * NNODES + n] + EPSV;
  float a = prelu_a[p];
  v4f o = ((v4f*)emb)[gid];
  #pragma unroll
  for (int i = 0; i < 4; ++i) {
    float t = o[i] / d;
    o[i] = t >= 0.f ? t : a * t;
  }
  ((v4f*)emb)[gid] = o;
}

// ---------------- GEMM 2: column sums of tanh(emb @ fc_w^T + b) ----------------
__global__ void __launch_bounds__(256) k_gemm_att(const float* __restrict__ emb,
                                                  const unsigned short* __restrict__ fcwbf,
                                                  const float* __restrict__ fcb,
                                                  float* __restrict__ sp) {
  extern __shared__ unsigned short smem[];
  const int tid = threadIdx.x;
  stage_B_async(smem, fcwbf, tid);

  const int lane = tid & 31;
  const int wave = tid >> 5;
  const int hl   = lane >> 4;
  const int l16  = lane & 15;
  const int rowBase = blockIdx.x * 64 + (wave & 3) * 16;
  const int jBase   = (wave >> 2) * 128;
  const int r = rowBase + l16;

  v8f C[8];
  #pragma unroll
  for (int t = 0; t < 8; ++t) __builtin_memset(&C[t], 0, sizeof(v8f));

  for (int kc = 0; kc < HDIM; kc += 32) {
    v16bf a;
    if (r < NNODES) {
      const float* ap = emb + (size_t)r * HDIM + kc + hl * 8;
      v4f f0 = *(const v4f*)(ap + 0);
      v4f f1 = *(const v4f*)(ap + 4);
      v4f f2 = *(const v4f*)(ap + 16);
      v4f f3 = *(const v4f*)(ap + 20);
      float tmp[16] = {f0[0],f0[1],f0[2],f0[3], f1[0],f1[1],f1[2],f1[3],
                       f2[0],f2[1],f2[2],f2[3], f3[0],f3[1],f3[2],f3[3]};
      #pragma unroll
      for (int i = 0; i < 16; ++i)
        a[i] = __builtin_bit_cast(__bf16, f2bf_raw(tmp[i]));
    } else {
      __builtin_memset(&a, 0, sizeof(a));
    }
    #pragma unroll
    for (int jt = 0; jt < 8; ++jt) {
      const unsigned short* bp = smem + (jBase + jt * 16 + l16) * BPITCH + kc + hl * 8;
      v8bf blo = *(const v8bf*)bp;
      v8bf bhi = *(const v8bf*)(bp + 16);
      v16bf b = __builtin_shufflevector(blo, bhi, 0,1,2,3,4,5,6,7,8,9,10,11,12,13,14,15);
      C[jt] = __builtin_amdgcn_wmma_f32_16x16x32_bf16(false, a, false, b,
                                                      (short)0, C[jt], false, false);
    }
  }

  // Fused tanh + column reduction; only valid rows contribute.
  #pragma unroll
  for (int jt = 0; jt < 8; ++jt) {
    const int col = jBase + jt * 16 + l16;
    const float bias = fcb[col];
    float s = 0.f;
    #pragma unroll
    for (int v = 0; v < 8; ++v) {
      int rr = rowBase + v + hl * 8;
      if (rr < NNODES) s += tanhf(C[jt][v] + bias);
    }
    s += __shfl_xor(s, 16, 32);         // combine the two half-wave M-halves
    if (hl == 0) global_fadd(&sp[col], s);
  }
}

// ---------------- beta = softmax((sp/N) @ att) ----------------
__global__ void __launch_bounds__(128) k_beta(const float* __restrict__ sp,
                                              const float* __restrict__ att,
                                              float* __restrict__ beta) {
  int tid = threadIdx.x;
  int p = tid >> 5, lane = tid & 31;
  float s = 0.f;
  for (int i = lane; i < HDIM; i += 32) s += sp[p * HDIM + i] * att[i];
  #pragma unroll
  for (int off = 16; off; off >>= 1) s += __shfl_xor(s, off, 32);
  __shared__ float lg[NPATH];
  if (lane == 0) lg[p] = s / (float)NNODES;
  __syncthreads();
  if (tid == 0) {
    float m = lg[0];
    for (int i = 1; i < NPATH; ++i) m = fmaxf(m, lg[i]);
    float e[NPATH], sum = 0.f;
    for (int i = 0; i < NPATH; ++i) { e[i] = __expf(lg[i] - m); sum += e[i]; }
    for (int i = 0; i < NPATH; ++i) beta[i] = e[i] / sum;
  }
}

// ---------------- z = sum_p beta[p] * emb[p] ----------------
__global__ void __launch_bounds__(256) k_combine(const float* __restrict__ emb,
                                                 const float* __restrict__ beta,
                                                 float* __restrict__ out) {
  long gid = (long)blockIdx.x * 256 + threadIdx.x;
  const long n4 = (long)NNODES * HDIM / 4;
  if (gid >= n4) return;
  float b0 = beta[0], b1 = beta[1], b2 = beta[2], b3 = beta[3];
  v4f a = ((const v4f*)emb)[gid];
  v4f b = ((const v4f*)emb)[gid + n4];
  v4f c = ((const v4f*)emb)[gid + 2 * n4];
  v4f d = ((const v4f*)emb)[gid + 3 * n4];
  v4f o;
  #pragma unroll
  for (int i = 0; i < 4; ++i) o[i] = b0 * a[i] + b1 * b[i] + b2 * c[i] + b3 * d[i];
  ((v4f*)out)[gid] = o;
}

// ---------------- host side ----------------
extern "C" void kernel_launch(void* const* d_in, const int* in_sizes, int n_in,
                              void* d_out, int out_size, void* d_ws, size_t ws_size,
                              hipStream_t stream) {
  const float* h    = (const float*)d_in[0];
  const int*   rows = (const int*)d_in[1];
  const int*   cols = (const int*)d_in[2];
  const float* vals = (const float*)d_in[3];
  const float* W    = (const float*)d_in[4];
  const float* pa   = (const float*)d_in[5];
  const float* fcw  = (const float*)d_in[6];
  const float* fcb  = (const float*)d_in[7];
  const float* att  = (const float*)d_in[8];
  (void)in_sizes; (void)n_in; (void)out_size; (void)ws_size;

  // Workspace carve-up (256B aligned). ~258 MB total.
  char* ws = (char*)d_ws;
  size_t off = 0;
  auto carve = [&](size_t bytes) -> void* {
    void* p = ws + off;
    off += (bytes + 255) & ~(size_t)255;
    return p;
  };
  unsigned short* hbf   = (unsigned short*)carve((size_t)NNODES * HDIM * 2);
  unsigned short* wbf   = (unsigned short*)carve((size_t)NPATH * HDIM * HDIM * 2);
  unsigned short* fcwbf = (unsigned short*)carve((size_t)HDIM * HDIM * 2);
  unsigned short* X16   = (unsigned short*)carve((size_t)NNODES * HDIM * 2); // reused per path
  float* EMB  = (float*)carve((size_t)NPATH * NNODES * HDIM * 4);
  float* deg  = (float*)carve((size_t)NPATH * NNODES * 4);
  float* sp   = (float*)carve((size_t)NPATH * HDIM * 4);
  float* beta = (float*)carve((size_t)NPATH * 4);

  // 1) zero accumulators
  {
    int n4 = NPATH * NNODES * HDIM / 4;            // 12.8M
    k_zero_f4<<<(n4 + 255) / 256, 256, 0, stream>>>(EMB, n4);
  }
  {
    int n4 = NPATH * NNODES / 4;
    k_zero_f4<<<(n4 + 255) / 256, 256, 0, stream>>>(deg, n4);
  }
  k_zero_f4<<<1, 256, 0, stream>>>(sp, NPATH * HDIM / 4);

  // 2) bf16 converts
  {
    int n4 = NNODES * HDIM / 4;
    k_f32_to_bf16<<<(n4 + 255) / 256, 256, 0, stream>>>(h, hbf, n4);
  }
  {
    int n4 = NPATH * HDIM * HDIM / 4;
    k_f32_to_bf16<<<(n4 + 255) / 256, 256, 0, stream>>>(W, wbf, n4);
  }
  {
    int n4 = HDIM * HDIM / 4;
    k_f32_to_bf16<<<(n4 + 255) / 256, 256, 0, stream>>>(fcw, fcwbf, n4);
  }

  const int rowBlocks = (NNODES + 63) / 64;
  const size_t ldsB = (size_t)HDIM * BPITCH * 2;   // 135168 B

  // 3) per-path: WMMA GEMM then edge scatter (stream order lets X16 be reused)
  for (int p = 0; p < NPATH; ++p) {
    k_gemm_xw<<<rowBlocks, 256, ldsB, stream>>>(hbf, wbf + (size_t)p * HDIM * HDIM, X16);
    long sthreads = (long)NEDGE * 64;
    k_scatter<<<(int)((sthreads + 255) / 256), 256, 0, stream>>>(
        rows + (size_t)p * NEDGE, cols + (size_t)p * NEDGE, vals + (size_t)p * NEDGE,
        X16, EMB + (size_t)p * NNODES * HDIM, deg + (size_t)p * NNODES);
  }

  // 4) normalize + PReLU
  {
    long n4 = (long)NPATH * NNODES * HDIM / 4;
    k_norm_prelu<<<(int)((n4 + 255) / 256), 256, 0, stream>>>(EMB, deg, pa);
  }

  // 5) attention GEMM with fused tanh + column sums
  for (int p = 0; p < NPATH; ++p)
    k_gemm_att<<<rowBlocks, 256, ldsB, stream>>>(EMB + (size_t)p * NNODES * HDIM,
                                                 fcwbf, fcb, sp + p * HDIM);

  // 6) softmax over metapaths
  k_beta<<<1, 128, 0, stream>>>(sp, att, beta);

  // 7) weighted combine into output
  {
    long n4 = (long)NNODES * HDIM / 4;
    k_combine<<<(int)((n4 + 255) / 256), 256, 0, stream>>>(EMB, beta, (float*)d_out);
  }
}
